// Net_Without_GINEgPool_81973745811909
// MI455X (gfx1250) — matrix operand order repaired and checked by reading, not compiled, not run on hardware
//
#include <hip/hip_runtime.h>
#include <math.h>

// ---------------------------------------------------------------------------
// Net_Without_GINEgPool — GATConv(32->128,heads=4) + TopKPooling(0.8) +
// max|mean pool + MLP head, for MI455X (gfx1250, wave32, WMMA).
//
// Stage plan:
//   1. pack W (32x512 f32) into WMMA B-fragment layout (f16, 32 KB)
//   2. h = x @ W via v_wmma_f32_16x16x32_f16  (K=32 == one WMMA per tile)
//   3. a_s/a_d per node (wave-per-node reduction)
//   4. edge pass1: e = leaky_relu(a_s[src]+a_d[dst]); segment-max via
//      sortable-uint atomicMax (correct for negative floats)
//   5. edge pass2: ex = exp(e - m[dst]); segment-sum via atomicAdd
//   6. edge pass3: agg[dst] += alpha * h[src]  (wave-per-edge, 512 cols)
//   7. x1 = relu(agg + bias) in-place; score = tanh(x1.w / ||w||)
//   8. per-graph Top-800 by rank counting (order-invariant downstream:
//      only max/mean pooling consumes the selection)
//   9. MLP 1024->128->64->8 + log_softmax
// ---------------------------------------------------------------------------

typedef __attribute__((ext_vector_type(16))) _Float16 v16h;
typedef __attribute__((ext_vector_type(8)))  float    v8f;

#define NNODES 32000
#define NPG    1000
#define NB     32
#define HD     512
#define FIN    32
#define KKEEP  800
#define MTILES (NNODES / 16)   // 2000
#define NTILES (HD / 16)       // 32

// ---------------- helpers ----------------
__device__ inline float wave_sum32(float v) {
#pragma unroll
  for (int off = 16; off > 0; off >>= 1) v += __shfl_xor(v, off, 32);
  return v;
}

// order-preserving f32 <-> u32 mapping so atomicMax(u32) == float max
__device__ inline unsigned f2sort(float v) {
  unsigned b = __float_as_uint(v);
  return (b & 0x80000000u) ? ~b : (b | 0x80000000u);
}
__device__ inline float sort2f(unsigned u) {
  return (u & 0x80000000u) ? __uint_as_float(u & 0x7FFFFFFFu)
                           : __uint_as_float(~u);
}

// ---------------- 1. pack W into B-fragment layout ----------------
// B (32x16, f16) fragment per n-tile t: lane 0-15 -> N=lane, K=0..15 in the
// 16 packed halves; lane 16-31 -> N=lane-16, K=16..31. Flat half index j maps
// to K = kb + j (kb = 0 or 16).  Wp[(t*32 + lane)*16 + j].
__global__ void pack_w_kernel(const float* __restrict__ W,
                              _Float16* __restrict__ Wp) {
  int tid  = threadIdx.x;            // 0..1023 == 32 tiles * 32 lanes
  int t    = tid >> 5, lane = tid & 31;
  int kb   = (lane & 16) ? 16 : 0;
  int n    = t * 16 + (lane & 15);
#pragma unroll
  for (int j = 0; j < 16; ++j)
    Wp[tid * 16 + j] = (_Float16)W[(kb + j) * HD + n];
}

// ---------------- 2. h = x @ W (WMMA) ----------------
// A (16x32 f16) layout (contiguous, matches Wp): lane<16 -> halves 0..15 = K0..15
//                               lane>=16 -> halves 0..15 = K16..31
// C/D (16x16 f32): VGPR r, lane l -> M = r + (l>=16 ? 8:0), N = l&15.
__global__ void gemm_xw_kernel(const float* __restrict__ x,
                               const _Float16* __restrict__ Wp,
                               float* __restrict__ h) {
  int lane = threadIdx.x & 31;
  int tile = blockIdx.x * 8 + (threadIdx.x >> 5);
  if (tile >= MTILES) return;

  int m  = tile * 16 + (lane & 15);
  int k0 = (lane & 16) ? 16 : 0;
  const float* xr = x + (size_t)m * FIN;
  v16h a;
#pragma unroll
  for (int j = 0; j < 8; ++j) {
    a[j]     = (_Float16)xr[k0 + j];
    a[j + 8] = (_Float16)xr[k0 + 8 + j];
  }

  int mc = tile * 16 + ((lane & 16) ? 8 : 0);
  int nc = lane & 15;
  for (int t = 0; t < NTILES; ++t) {
    v16h b = *(const v16h*)(Wp + ((size_t)(t * 32 + lane)) * 16);
    v8f  c = {};
    c = __builtin_amdgcn_wmma_f32_16x16x32_f16(false, a, false, b,
                                               (short)0, c, false, false);
#pragma unroll
    for (int r = 0; r < 8; ++r)
      h[(size_t)(mc + r) * HD + t * 16 + nc] = c[r];
  }
}

// ---------------- 3. a_s / a_d per node ----------------
// c = lane + 32*i  ->  head = i>>2 (compile-time after unroll)
__global__ void attn_ab_kernel(const float* __restrict__ h,
                               const float* __restrict__ att_src,
                               const float* __restrict__ att_dst,
                               float* __restrict__ a_s,
                               float* __restrict__ a_d) {
  int lane = threadIdx.x & 31;
  int n = blockIdx.x * 8 + (threadIdx.x >> 5);
  if (n >= NNODES) return;
  float sa[4] = {0.f, 0.f, 0.f, 0.f}, sd[4] = {0.f, 0.f, 0.f, 0.f};
#pragma unroll
  for (int i = 0; i < 16; ++i) {
    int c = lane + 32 * i;
    float hv = h[(size_t)n * HD + c];
    sa[i >> 2] += hv * att_src[c];
    sd[i >> 2] += hv * att_dst[c];
  }
#pragma unroll
  for (int hh = 0; hh < 4; ++hh) {
    float ra = wave_sum32(sa[hh]);
    float rd = wave_sum32(sd[hh]);
    if (lane == 0) { a_s[n * 4 + hh] = ra; a_d[n * 4 + hh] = rd; }
  }
}

// ---------------- 4. edge pass 1: e + segment max ----------------
__global__ void edge_pass1_kernel(const long long* __restrict__ src,
                                  const long long* __restrict__ dst,
                                  const float* __restrict__ a_s,
                                  const float* __restrict__ a_d,
                                  float* __restrict__ ebuf,
                                  unsigned* __restrict__ mbuf,
                                  int E, int E2) {
  int e = blockIdx.x * blockDim.x + threadIdx.x;
  if (e >= E2) return;
  int s, d;
  if (e < E) { s = (int)src[e]; d = (int)dst[e]; }
  else       { s = d = e - E; }                    // self loop
#pragma unroll
  for (int hh = 0; hh < 4; ++hh) {
    float v = a_s[s * 4 + hh] + a_d[d * 4 + hh];
    v = (v > 0.f) ? v : 0.2f * v;                  // leaky_relu slope 0.2
    ebuf[(size_t)e * 4 + hh] = v;
    atomicMax(&mbuf[d * 4 + hh], f2sort(v));
  }
}

// ---------------- 5. edge pass 2: exp + segment sum ----------------
__global__ void edge_pass2_kernel(const long long* __restrict__ src,
                                  const long long* __restrict__ dst,
                                  float* __restrict__ ebuf,
                                  const unsigned* __restrict__ mbuf,
                                  float* __restrict__ denom,
                                  int E, int E2) {
  int e = blockIdx.x * blockDim.x + threadIdx.x;
  if (e >= E2) return;
  int d = (e < E) ? (int)dst[e] : (e - E);
#pragma unroll
  for (int hh = 0; hh < 4; ++hh) {
    float m  = sort2f(mbuf[d * 4 + hh]);
    float ex = expf(ebuf[(size_t)e * 4 + hh] - m);
    ebuf[(size_t)e * 4 + hh] = ex;
    atomicAdd(&denom[d * 4 + hh], ex);
  }
}

// ---------------- 6. edge pass 3: agg[dst] += alpha * h[src] ----------------
__global__ void edge_pass3_kernel(const long long* __restrict__ src,
                                  const long long* __restrict__ dst,
                                  const float* __restrict__ ebuf,
                                  const float* __restrict__ denom,
                                  const float* __restrict__ h,
                                  float* __restrict__ agg,
                                  int E, int E2) {
  int lane = threadIdx.x & 31;
  int e = blockIdx.x * 8 + (threadIdx.x >> 5);
  if (e >= E2) return;
  int s, d;
  if (e < E) { s = (int)src[e]; d = (int)dst[e]; }
  else       { s = d = e - E; }
  float al[4];
#pragma unroll
  for (int hh = 0; hh < 4; ++hh)
    al[hh] = ebuf[(size_t)e * 4 + hh] / denom[d * 4 + hh];
#pragma unroll
  for (int i = 0; i < 16; ++i) {
    int c = lane + 32 * i;                          // head = i>>2
    atomicAdd(&agg[(size_t)d * HD + c], al[i >> 2] * h[(size_t)s * HD + c]);
  }
}

// ---------------- 7. x1 = relu(agg+bias); score = tanh(x1.w/||w||) ----------
__global__ void node_score_kernel(float* __restrict__ x1,   // agg in-place
                                  const float* __restrict__ bias,
                                  const float* __restrict__ tw,
                                  float* __restrict__ score) {
  int lane = threadIdx.x & 31;
  int n = blockIdx.x * 8 + (threadIdx.x >> 5);
  if (n >= NNODES) return;
  float dot = 0.f, wn = 0.f;
#pragma unroll
  for (int i = 0; i < 16; ++i) {
    int c = lane + 32 * i;
    float v = x1[(size_t)n * HD + c] + bias[c];
    v = fmaxf(v, 0.f);
    x1[(size_t)n * HD + c] = v;
    float w = tw[c];
    dot += v * w;
    wn  += w * w;
  }
  dot = wave_sum32(dot);
  wn  = wave_sum32(wn);
  if (lane == 0) score[n] = tanhf(dot * rsqrtf(wn));
}

// ---------------- 8. per-graph Top-800 + max|mean pooling ----------------
__global__ void topk_pool_kernel(const float* __restrict__ x1,
                                 const float* __restrict__ score,
                                 float* __restrict__ x2) {
  __shared__ float s_sc[NPG];
  __shared__ int   s_kept[NPG];
  int b = blockIdx.x, tid = threadIdx.x;
  if (tid < NPG) s_sc[tid] = score[b * NPG + tid];
  __syncthreads();
  if (tid < NPG) {
    float si = s_sc[tid];
    int r = 0;
    for (int j = 0; j < NPG; ++j) {
      float sj = s_sc[j];
      r += (sj > si) || (sj == si && j < tid);   // top_k tie-break: low index
    }
    s_kept[tid] = (r < KKEEP);
  }
  __syncthreads();
  if (tid < HD) {
    float mx = -1e30f, sm = 0.f;
    for (int nn = 0; nn < NPG; ++nn) {
      if (s_kept[nn]) {
        float v = x1[((size_t)(b * NPG + nn)) * HD + tid] * s_sc[nn];
        mx = fmaxf(mx, v);
        sm += v;
      }
    }
    x2[b * 1024 + tid]      = mx;
    x2[b * 1024 + HD + tid] = sm * (1.0f / KKEEP);
  }
}

// ---------------- 9. MLP head + log_softmax ----------------
__global__ void mlp_kernel(const float* __restrict__ x2,
                           const float* __restrict__ w1, const float* __restrict__ b1,
                           const float* __restrict__ w2, const float* __restrict__ b2,
                           const float* __restrict__ w3, const float* __restrict__ b3,
                           float* __restrict__ out) {
  __shared__ float xl[1024];
  __shared__ float h1[128];
  __shared__ float h2[64];
  __shared__ float lg[8];
  int b = blockIdx.x, tid = threadIdx.x;   // blockDim = 128
  for (int k = tid; k < 1024; k += 128) xl[k] = x2[b * 1024 + k];
  __syncthreads();
  {
    float acc = b1[tid];
    for (int k = 0; k < 1024; ++k) acc += xl[k] * w1[k * 128 + tid];
    h1[tid] = fmaxf(acc, 0.f);
  }
  __syncthreads();
  if (tid < 64) {
    float acc = b2[tid];
    for (int k = 0; k < 128; ++k) acc += h1[k] * w2[k * 64 + tid];
    h2[tid] = fmaxf(acc, 0.f);
  }
  __syncthreads();
  if (tid < 8) {
    float acc = b3[tid];
    for (int k = 0; k < 64; ++k) acc += h2[k] * w3[k * 8 + tid];
    lg[tid] = acc;
  }
  __syncthreads();
  if (tid == 0) {
    float mx = lg[0];
    for (int j = 1; j < 8; ++j) mx = fmaxf(mx, lg[j]);
    float s = 0.f;
    for (int j = 0; j < 8; ++j) s += expf(lg[j] - mx);
    float lse = mx + logf(s);
    for (int j = 0; j < 8; ++j) out[b * 8 + j] = lg[j] - lse;
  }
}

// ---------------------------------------------------------------------------
extern "C" void kernel_launch(void* const* d_in, const int* in_sizes, int n_in,
                              void* d_out, int out_size, void* d_ws, size_t ws_size,
                              hipStream_t stream) {
  const float*     x     = (const float*)d_in[0];
  const long long* ei    = (const long long*)d_in[1];   // int64 [2,E]
  // d_in[2] = batch (unused), d_in[3] = edge_attr (unused by reference)
  const float* W     = (const float*)d_in[4];
  const float* att_s = (const float*)d_in[5];
  const float* att_d = (const float*)d_in[6];
  const float* bias  = (const float*)d_in[7];
  const float* tw    = (const float*)d_in[8];
  const float* w1 = (const float*)d_in[9];   const float* b1 = (const float*)d_in[10];
  const float* w2 = (const float*)d_in[11];  const float* b2 = (const float*)d_in[12];
  const float* w3 = (const float*)d_in[13];  const float* b3 = (const float*)d_in[14];
  float* out = (float*)d_out;

  const int E  = in_sizes[1] / 2;           // 320000
  const int E2 = E + NNODES;                // + self loops
  const long long* srcp = ei;
  const long long* dstp = ei + E;

  // workspace carve-up (all chunks 256B-aligned)
  char* ws = (char*)d_ws;
  size_t off = 0;
  float*    h_buf = (float*)(ws + off);    off += (size_t)NNODES * HD * 4;  // 65.5 MB
  float*    agg   = (float*)(ws + off);    off += (size_t)NNODES * HD * 4;  // 65.5 MB
  _Float16* Wp    = (_Float16*)(ws + off); off += (size_t)32 * 512 * 2;     // 32 KB
  float*    a_sb  = (float*)(ws + off);    off += (size_t)NNODES * 4 * 4;
  float*    a_db  = (float*)(ws + off);    off += (size_t)NNODES * 4 * 4;
  unsigned* mb    = (unsigned*)(ws + off); off += (size_t)NNODES * 4 * 4;
  float*    den   = (float*)(ws + off);    off += (size_t)NNODES * 4 * 4;
  float*    ebuf  = (float*)(ws + off);    off += (size_t)E2 * 4 * 4;       // 5.6 MB
  float*    score = (float*)(ws + off);    off += (size_t)NNODES * 4;
  float*    x2    = (float*)(ws + off);    off += (size_t)NB * 1024 * 4;

  hipMemsetAsync(agg, 0, (size_t)NNODES * HD * 4, stream);
  hipMemsetAsync(mb,  0, (size_t)NNODES * 4 * 4, stream);  // 0 < sortable(-inf)
  hipMemsetAsync(den, 0, (size_t)NNODES * 4 * 4, stream);

  pack_w_kernel <<<1, 1024, 0, stream>>>(W, Wp);
  gemm_xw_kernel<<<MTILES / 8, 256, 0, stream>>>(x, Wp, h_buf);
  attn_ab_kernel<<<NNODES / 8, 256, 0, stream>>>(h_buf, att_s, att_d, a_sb, a_db);
  edge_pass1_kernel<<<(E2 + 255) / 256, 256, 0, stream>>>(srcp, dstp, a_sb, a_db,
                                                          ebuf, mb, E, E2);
  edge_pass2_kernel<<<(E2 + 255) / 256, 256, 0, stream>>>(srcp, dstp, ebuf, mb,
                                                          den, E, E2);
  edge_pass3_kernel<<<(E2 + 7) / 8, 256, 0, stream>>>(srcp, dstp, ebuf, den,
                                                      h_buf, agg, E, E2);
  node_score_kernel<<<NNODES / 8, 256, 0, stream>>>(agg, bias, tw, score);
  topk_pool_kernel<<<NB, 1024, 0, stream>>>(agg, score, x2);
  mlp_kernel<<<NB, 128, 0, stream>>>(x2, w1, b1, w2, b2, w3, b3, out);
}
